// FeatureAggregationModule_1949915152906
// MI455X (gfx1250) — matrix operand order repaired and checked by reading, not compiled, not run on hardware
//
#include <hip/hip_runtime.h>

// ---------------------------------------------------------------------------
// MI455X (gfx1250) implementation.
// Convs (7.7e10 FLOPs) are compute-bound -> bf16 WMMA implicit GEMM.
// Attention (~6e8 FLOPs) is bandwidth-trivial -> f32 VALU.
// Conv: each workgroup computes a 128(cout) x 64(pixel) half-row tile.
// Per wave: 4 acc tiles + 4 B fragments + A double-buffer (~90 live VGPRs),
// below the allocator's pressure cliff -> distinct B tuples (batched
// ds_load_b128 clause, staggered dscnt waits) AND in-place WMMA accumulation.
// ---------------------------------------------------------------------------

typedef __attribute__((ext_vector_type(16))) __bf16 v16bf;
typedef __attribute__((ext_vector_type(8)))  __bf16 v8bf;
typedef __attribute__((ext_vector_type(8)))  float  v8f;
typedef unsigned short u16_t;
typedef __attribute__((ext_vector_type(8)))  u16_t  us8;

#define BATCH 4
#define CCH   128         // channels in == out
#define HW    128         // H == W
#define KDIM  (9*CCH)     // GEMM K = 1152, ordered K = (r*3+s)*128 + ci
#define NKCHUNK (KDIM/32) // 36 chunks of K=32
#define NPIX  64          // pixels per workgroup (half row)
#define LCOLS (NPIX+2)    // 66 staged cols: halo col each side
#define LCPAD 136         // channel-dim pad (272B stride -> rotates LDS banks)
#define LDSBYTES (3*LCOLS*LCPAD*2)   // 53,856 B

// ---------------------------------------------------------------------------
// f32 -> bf16 activation convert
// ---------------------------------------------------------------------------
__global__ __launch_bounds__(256)
void f32_to_bf16(const float* __restrict__ src, __bf16* __restrict__ dst, int n) {
  int i = blockIdx.x * 256 + threadIdx.x;
  if (i < n) dst[i] = (__bf16)src[i];
}

// OIHW f32 weight -> bf16 GEMM-A rows: wbf[co*1152 + (r*3+s)*128 + ci]
__global__ __launch_bounds__(256)
void cvt_weight(const float* __restrict__ w, __bf16* __restrict__ wbf) {
  int i = blockIdx.x * 256 + threadIdx.x;           // over 128*128*9
  if (i < CCH * CCH * 9) {
    int s  = i % 3;
    int r  = (i / 3) % 3;
    int ci = (i / 9) % CCH;
    int co = i / (9 * CCH);
    wbf[(size_t)co * KDIM + (r * 3 + s) * CCH + ci] = (__bf16)w[i];
  }
}

// ---------------------------------------------------------------------------
// conv3x3 (SAME, bias) as implicit GEMM via V_WMMA_F32_16X16X32_BF16.
// grid.x = B*H*2 (one 64-pixel half-row per workgroup); block = 256 (8 waves).
// Wave `wid` owns cout rows [16*wid, 16*wid+16); loops 4 pixel tiles of 16.
// ---------------------------------------------------------------------------
__global__ __launch_bounds__(256)
void conv3x3_wmma_bf16(const __bf16* __restrict__ src,   // [B][C][H][W] bf16
                       const __bf16* __restrict__ wbf,   // [Cout][KDIM] bf16
                       const float*  __restrict__ bias,  // [Cout]
                       float* __restrict__ dst)          // [B][C][H][W] f32
{
  extern __shared__ __bf16 lds[];   // [rr=3][col=66][ci=136(pad)]
  const int tid   = threadIdx.x;
  const int b     = blockIdx.x >> 8;
  const int rem   = blockIdx.x & 255;
  const int h     = rem >> 1;
  const int side  = rem & 1;
  const int wbase = side * NPIX;

  // ---- stage rows h-1..h+1, cols wbase-1..wbase+64, all channels ----
  // LDS col = w - wbase + 1 ; channel-contiguous layout for B fragments.
  for (int idx = tid; idx < 3 * CCH; idx += 256) {
    const int rr = idx >> 7;          // 0..2
    const int c  = idx & (CCH - 1);
    const int gh = h + rr - 1;
    const bool valid = (gh >= 0) && (gh < HW);
    const int ghc = valid ? gh : 0;
    const __bf16* grow = src + (((size_t)b * CCH + c) * HW + ghc) * HW;
    const us8* gv = (const us8*)(grow + wbase);    // 16B-aligned (wbase 0/64)
    u16_t* baseu = (u16_t*)lds + (size_t)rr * LCOLS * LCPAD + c;
    #pragma unroll 4
    for (int w8 = 0; w8 < NPIX / 8; ++w8) {
      us8 d = {};
      if (valid) d = gv[w8];                       // coalesced 16B global load
      #pragma unroll
      for (int j = 0; j < 8; ++j)
        baseu[(size_t)(w8 * 8 + j + 1) * LCPAD] = d[j];
    }
    // halo columns (zero at image border)
    u16_t le = 0, re = 0;
    if (valid && wbase > 0)          le = *(const u16_t*)(grow + wbase - 1);
    if (valid && wbase + NPIX < HW)  re = *(const u16_t*)(grow + wbase + NPIX);
    baseu[0] = le;
    baseu[(size_t)(NPIX + 1) * LCPAD] = re;
  }
  __syncthreads();

  const int wid  = tid >> 5;      // wave id 0..7
  const int lane = tid & 31;
  const int n    = lane & 15;     // A row / B,C,D column within tile
  const int hf   = lane >> 4;     // half-wave select
  const int M0   = wid << 4;

  // C/D layout: lane n = N, VGPR r -> M = r + 8*hf. Init acc with bias.
  float binit[8];
  #pragma unroll
  for (int r = 0; r < 8; ++r) binit[r] = bias[M0 + r + 8 * hf];
  v8f acc[4];
  #pragma unroll
  for (int nt = 0; nt < 4; ++nt) {
    #pragma unroll
    for (int r = 0; r < 8; ++r) acc[nt][r] = binit[r];
  }

  // A layout (16-bit 16x32): lane row = lane&15; elem e=2v+j has
  //   K = kk*32 + 16*hf + e
  const __bf16* wrow = wbf + (size_t)(M0 + n) * KDIM;

  auto load_afrag = [&](int kk) -> v16bf {
    v16bf a;
    #pragma unroll
    for (int vvv = 0; vvv < 8; ++vvv) {
      const int t = (hf << 4) + (vvv << 1);
      const __bf16* ap = wrow + kk * 32 + t;       // merges to 2x global b128
      a[2 * vvv]     = ap[0];
      a[2 * vvv + 1] = ap[1];
    }
    return a;
  };

  // B layout (16-bit 32x16): lane col = lane&15; elem e has K = 16*hf + e
  //   -> ci = cib + 16*hf + e : contiguous in LDS channel dim.
  auto load_bfrag = [&](const __bf16* lrow, int col) -> v16bf {
    const __bf16* bp = lrow + (size_t)col * LCPAD;
    v8bf lo = *(const v8bf*)bp;                    // ds_load_b128
    v8bf hi = *(const v8bf*)(bp + 8);              // ds_load_b128
    return __builtin_shufflevector(lo, hi,
        0, 1, 2, 3, 4, 5, 6, 7, 8, 9, 10, 11, 12, 13, 14, 15);
  };

  v16bf a = load_afrag(0);

  for (int kk = 0; kk < NKCHUNK; ++kk) {           // 36 K-chunks of 32
    const int rs  = kk >> 2;                       // tap index 0..8 (128%32==0)
    const int cib = (kk & 3) << 5;                 // ci base within chunk
    const int rr  = rs / 3;                        // vertical tap
    const int sc  = rs - rr * 3;                   // horizontal tap
    const __bf16* lrow = lds + (size_t)rr * LCOLS * LCPAD + cib + (hf << 4);

    // Load ALL 4 B fragments first: 8-deep ds_load_b128 clause, then the
    // WMMA burst consumes them under staggered dscnt waits.
    v16bf bfr[4];
    #pragma unroll
    for (int nt = 0; nt < 4; ++nt)
      bfr[nt] = load_bfrag(lrow, (nt << 4) + n + sc);

    // Prefetch next A fragment into registers across the WMMA burst.
    v16bf a_next = a;
    if (kk + 1 < NKCHUNK) a_next = load_afrag(kk + 1);

    #pragma unroll
    for (int nt = 0; nt < 4; ++nt)
      acc[nt] = __builtin_amdgcn_wmma_f32_16x16x32_bf16(
          false, a, false, bfr[nt], (short)0, acc[nt], false, false);

    a = a_next;
  }

  #pragma unroll
  for (int nt = 0; nt < 4; ++nt) {
    const int w = wbase + (nt << 4) + n;
    #pragma unroll
    for (int r = 0; r < 8; ++r) {
      const int cout = M0 + r + 8 * hf;
      dst[(((size_t)b * CCH + cout) * HW + h) * HW + w] = acc[nt][r];
    }
  }
}

// ---------------------------------------------------------------------------
// Local-window attention + masked aggregation + final sum (bandwidth-light).
// block = 256 threads = one 16x16 pixel tile; grid = B*8*8 = 256.
// ---------------------------------------------------------------------------
__global__ __launch_bounds__(256)
void fam_attention(const float* __restrict__ q,
                   const float* __restrict__ vconv,
                   const float* __restrict__ kb,
                   const float* __restrict__ kf,
                   const int*   __restrict__ mask,
                   float* __restrict__ out,
                   float* __restrict__ attb,
                   float* __restrict__ attf,
                   float* __restrict__ mout)
{
  const int blk = blockIdx.x;
  const int b  = blk >> 6;
  const int ty = (blk >> 3) & 7;
  const int tx = blk & 7;
  const int h = (ty << 4) + (threadIdx.x >> 4);
  const int w = (tx << 4) + (threadIdx.x & 15);

  // mask: [B][1][64][64] int, nearest-upsampled 2x
  const float maskf =
      mask[((size_t)b * 64 + (h >> 1)) * 64 + (w >> 1)] ? 1.0f : 0.0f;

  const long long plane = (long long)HW * HW;
  const long long pix   = (long long)h * HW + w;
  const long long base  = (long long)b * CCH * plane + pix;

  float db[9], df[9];
  #pragma unroll
  for (int p = 0; p < 9; ++p) { db[p] = 0.f; df[p] = 0.f; }

  // pass 1: q . k-patch dots over channels
  for (int c = 0; c < CCH; ++c) {
    const long long cb = base + (long long)c * plane;
    const float qv = q[cb];
    #pragma unroll
    for (int p = 0; p < 9; ++p) {
      const int dh = p / 3 - 1, dw = p % 3 - 1;
      const int hh = h + dh, ww = w + dw;
      const bool in = (hh >= 0) & (hh < HW) & (ww >= 0) & (ww < HW);
      const long long off = cb + (long long)dh * HW + dw;
      const float kbv = in ? kb[off] : 0.f;
      const float kfv = in ? kf[off] : 0.f;
      db[p] += qv * kbv;
      df[p] += qv * kfv;
    }
  }

  const float scale = 0.08838834764831845f;        // 1/sqrt(128)
  float mb = -1e30f, mf = -1e30f;
  #pragma unroll
  for (int p = 0; p < 9; ++p) {
    db[p] *= scale; df[p] *= scale;
    mb = fmaxf(mb, db[p]); mf = fmaxf(mf, df[p]);
  }
  float ab[9], af[9], sb = 0.f, sf = 0.f;
  #pragma unroll
  for (int p = 0; p < 9; ++p) {
    ab[p] = __expf(db[p] - mb); sb += ab[p];
    af[p] = __expf(df[p] - mf); sf += af[p];
  }
  const float rb = 1.f / sb, rf = 1.f / sf;
  #pragma unroll
  for (int p = 0; p < 9; ++p) {
    ab[p] *= rb; af[p] *= rf;
    attb[((long long)b * 9 + p) * plane + pix] = db[p] * maskf;  // masked logits
    attf[((long long)b * 9 + p) * plane + pix] = df[p] * maskf;
  }

  // pass 2: aggregate key patches with softmax weights, add v, apply mask
  for (int c = 0; c < CCH; ++c) {
    const long long cb = base + (long long)c * plane;
    float sb2 = 0.f, sf2 = 0.f;
    #pragma unroll
    for (int p = 0; p < 9; ++p) {
      const int dh = p / 3 - 1, dw = p % 3 - 1;
      const int hh = h + dh, ww = w + dw;
      const bool in = (hh >= 0) & (hh < HW) & (ww >= 0) & (ww < HW);
      const long long off = cb + (long long)dh * HW + dw;
      const float kbv = in ? kb[off] : 0.f;
      const float kfv = in ? kf[off] : 0.f;
      sb2 += ab[p] * kbv;
      sf2 += af[p] * kfv;
    }
    out[cb] = vconv[cb] + maskf * (sb2 + sf2);
  }
  mout[(long long)b * plane + pix] = maskf;        // bool mask as 0/1
}

// ---------------------------------------------------------------------------
// Launch. Workspace layout (~185 MB):
//   bf16: x, b, f (16.8MB each), wq/wk/wv (0.29MB each)
//   f32 : q, v, kb, kf (33.5MB each)
// Output: [out 8388608 | attb 589824 | attf 589824 | mask 65536] floats.
// ---------------------------------------------------------------------------
extern "C" void kernel_launch(void* const* d_in, const int* in_sizes, int n_in,
                              void* d_out, int out_size, void* d_ws, size_t ws_size,
                              hipStream_t stream) {
  const float* x    = (const float*)d_in[0];
  const float* bt   = (const float*)d_in[1];
  const float* ft   = (const float*)d_in[2];
  const float* wq   = (const float*)d_in[3];
  const float* bq   = (const float*)d_in[4];
  const float* wk   = (const float*)d_in[5];
  const float* bk   = (const float*)d_in[6];
  const float* wvw  = (const float*)d_in[7];
  const float* bv   = (const float*)d_in[8];
  const int*   mask = (const int*)d_in[9];

  const size_t ACT = (size_t)BATCH * CCH * HW * HW;   // 8,388,608 elems
  const size_t WSZ = (size_t)CCH * KDIM;              // 147,456 elems

  __bf16* xb  = (__bf16*)d_ws;
  __bf16* bb  = xb + ACT;
  __bf16* fb  = bb + ACT;
  __bf16* wqb = fb + ACT;
  __bf16* wkb = wqb + WSZ;
  __bf16* wvb = wkb + WSZ;
  float*  qb  = (float*)(wvb + WSZ);
  float*  vb  = qb + ACT;
  float*  kbb = vb + ACT;
  float*  kfb = kbb + ACT;

  float* out   = (float*)d_out;
  float* attbp = out + ACT;
  float* attfp = attbp + (size_t)BATCH * 9 * HW * HW;
  float* moutp = attfp + (size_t)BATCH * 9 * HW * HW;

  dim3 blk(256);
  dim3 gact((unsigned)((ACT + 255) / 256));
  dim3 gw((unsigned)((WSZ + 255) / 256));

  f32_to_bf16<<<gact, blk, 0, stream>>>(x,  xb, (int)ACT);
  f32_to_bf16<<<gact, blk, 0, stream>>>(bt, bb, (int)ACT);
  f32_to_bf16<<<gact, blk, 0, stream>>>(ft, fb, (int)ACT);
  cvt_weight<<<gw, blk, 0, stream>>>(wq,  wqb);
  cvt_weight<<<gw, blk, 0, stream>>>(wk,  wkb);
  cvt_weight<<<gw, blk, 0, stream>>>(wvw, wvb);

  dim3 gconv(BATCH * HW * 2);   // 1024 workgroups: one half-row each
  conv3x3_wmma_bf16<<<gconv, blk, LDSBYTES, stream>>>(xb, wqb, bq, qb);
  conv3x3_wmma_bf16<<<gconv, blk, LDSBYTES, stream>>>(xb, wvb, bv, vb);
  conv3x3_wmma_bf16<<<gconv, blk, LDSBYTES, stream>>>(bb, wkb, bk, kbb);
  conv3x3_wmma_bf16<<<gconv, blk, LDSBYTES, stream>>>(fb, wkb, bk, kfb);

  fam_attention<<<dim3(BATCH * 64), blk, 0, stream>>>(
      qb, vb, kbb, kfb, mask, out, attbp, attfp, moutp);
}
